// SparseMoE_29721173688811
// MI455X (gfx1250) — matrix operand order repaired and checked by reading, not compiled, not run on hardware
//
#include <hip/hip_runtime.h>

#define IN_DIM   1024
#define OUT_DIM  1024
#define NUM_E    8
#define NTOK     8192

#define M_TILE   64
#define XSTRIDE  1032                  // bf16 elems per LDS row: 2064B = 129*16, bank-spread
#define TILES_PER_E (NTOK / M_TILE)    // 128
#define KSTEPS   (IN_DIM / 32)         // 32 k-steps of 32
#define NT_TILES (OUT_DIM / 16)        // 64 column tiles
#define WBLK     512                   // bf16 elems per (ntile,kstep) fragment block: 32 lanes * 16

typedef __bf16 bf16_t;
typedef __attribute__((ext_vector_type(16))) __bf16 v16bf;
typedef __attribute__((ext_vector_type(8)))  __bf16 bf16x8;
typedef __attribute__((ext_vector_type(8)))  float  v8f;
typedef __attribute__((ext_vector_type(4)))  float  f32x4;

#define CAT16(a, b) __builtin_shufflevector((a), (b), 0,1,2,3,4,5,6,7,8,9,10,11,12,13,14,15)

// ---------------- zero counters ----------------
__global__ void moe_zero_kernel(int* __restrict__ cnt) {
  if (threadIdx.x < NUM_E) cnt[threadIdx.x] = 0;
}

// ---------------- weight pre-swizzle: f32 [e][k][n] -> bf16 WMMA-B fragment order ----------------
// dst element (e, nt, kt, lane, i)  <-  src[e][kt*32 + (lane>>4)*16 + i][nt*16 + (lane&15)]
__global__ __launch_bounds__(256)
void moe_prep_w(const float* __restrict__ w1, const float* __restrict__ w2,
                bf16_t* __restrict__ w1bf, bf16_t* __restrict__ w2bf)
{
  const float* src = blockIdx.y ? w2 : w1;
  bf16_t*      dst = blockIdx.y ? w2bf : w1bf;
  const int t    = blockIdx.x * 256 + threadIdx.x;   // 0 .. 8*64*32*32-1
  const int lane = t & 31;
  const int kt   = (t >> 5) & 31;
  const int nt   = (t >> 10) & 63;
  const int e    = t >> 16;
  const int n  = nt * 16 + (lane & 15);
  const int k0 = kt * 32 + (lane >> 4) * 16;
  const float* s = src + (size_t)e * IN_DIM * OUT_DIM + (size_t)k0 * OUT_DIM + n;
  bf16_t tmp[16];
  #pragma unroll
  for (int i = 0; i < 16; ++i) tmp[i] = (bf16_t)s[(size_t)i * OUT_DIM];
  bf16x8* d = (bf16x8*)(dst + (size_t)t * 16);
  d[0] = *(bf16x8*)&tmp[0];
  d[1] = *(bf16x8*)&tmp[8];
}

// ---------------- router: one wave32 per token (+ x -> bf16 conversion) ----------------
__global__ __launch_bounds__(256)
void moe_router_kernel(const float* __restrict__ x,
                       const float* __restrict__ rw,
                       const float* __restrict__ rb,
                       float* __restrict__ fused_out,
                       float* __restrict__ gating_out,
                       bf16_t* __restrict__ xbf,
                       int*   __restrict__ cnt,
                       int*   __restrict__ lists,
                       float* __restrict__ gates)
{
  const int gt   = blockIdx.x * blockDim.x + threadIdx.x;
  const int tok  = gt >> 5;
  const int lane = gt & 31;
  if (tok >= NTOK) return;

  const float* xr = x + (size_t)tok * IN_DIM;
  bf16_t* xbr = xbf + (size_t)tok * IN_DIM;
  float acc[NUM_E];
  #pragma unroll
  for (int e = 0; e < NUM_E; ++e) acc[e] = 0.f;

  for (int k = lane; k < IN_DIM; k += 32) {
    const float xv = xr[k];
    xbr[k] = (bf16_t)xv;                       // bf16 copy of activations
    const f32x4 w0 = *(const f32x4*)(rw + (size_t)k * NUM_E);
    const f32x4 w1 = *(const f32x4*)(rw + (size_t)k * NUM_E + 4);
    acc[0] += xv * w0.x; acc[1] += xv * w0.y; acc[2] += xv * w0.z; acc[3] += xv * w0.w;
    acc[4] += xv * w1.x; acc[5] += xv * w1.y; acc[6] += xv * w1.z; acc[7] += xv * w1.w;
  }
  #pragma unroll
  for (int e = 0; e < NUM_E; ++e) {
    #pragma unroll
    for (int off = 16; off > 0; off >>= 1)
      acc[e] += __shfl_xor(acc[e], off, 32);
  }

  // zero the fused output row (accumulated later via atomics)
  float* orow = fused_out + (size_t)tok * OUT_DIM;
  const f32x4 z = {0.f, 0.f, 0.f, 0.f};
  for (int j = lane; j < OUT_DIM / 4; j += 32) ((f32x4*)orow)[j] = z;

  if (lane == 0) {
    float lg[NUM_E];
    #pragma unroll
    for (int e = 0; e < NUM_E; ++e) lg[e] = acc[e] + rb[e];
    int e1 = 0; float v1 = lg[0];
    #pragma unroll
    for (int e = 1; e < NUM_E; ++e) if (lg[e] > v1) { v1 = lg[e]; e1 = e; }
    int e2 = (e1 == 0) ? 1 : 0; float v2 = lg[e2];
    #pragma unroll
    for (int e = 0; e < NUM_E; ++e) if (e != e1 && lg[e] > v2) { v2 = lg[e]; e2 = e; }

    const float t  = __expf(v2 - v1);
    const float g1 = 1.f / (1.f + t);
    const float g2 = t  / (1.f + t);

    float* grow = gating_out + (size_t)tok * NUM_E;
    #pragma unroll
    for (int e = 0; e < NUM_E; ++e) grow[e] = 0.f;
    grow[e1] = g1; grow[e2] = g2;

    int p1 = atomicAdd(&cnt[e1], 1);
    lists[e1 * NTOK + p1] = tok; gates[e1 * NTOK + p1] = g1;
    int p2 = atomicAdd(&cnt[e2], 1);
    lists[e2 * NTOK + p2] = tok; gates[e2 * NTOK + p2] = g2;
  }
}

// ---------------- one FFN layer over a 64-row tile, 2x2 register blocking per wave ----------------
// A: LDS [M_TILE][1024] (stride XSTRIDE).  wb: pre-swizzled bf16 fragments [64 nt][32 kt][512].
// EPI==0: sH = relu(acc+bias).  EPI==1: out[tok] += gate*(acc+bias)  (atomic f32).
template<int EPI>
__device__ __forceinline__
void ffn_layer(const bf16_t* __restrict__ A, bf16_t* __restrict__ sH,
               const int* __restrict__ sTok, const float* __restrict__ sGate,
               const bf16_t* __restrict__ wb,
               const float* __restrict__ bias,
               float* __restrict__ out)
{
  const int tid  = threadIdx.x;
  const int lane = tid & 31;
  const int wv   = tid >> 5;
  const int m0   = (wv & 1) * 32;      // rows m0..m0+31 (two 16-row A tiles)
  const int cg   = wv >> 1;            // 0..3: 256-column quarter (16 col tiles)
  const int lhi  = lane >> 4;
  const int llo  = lane & 15;

  for (int p = 0; p < 8; ++p) {        // 8 pairs of column tiles per wave
    const int nt0 = cg * 16 + p * 2;
    const int nt1 = nt0 + 1;
    const bf16_t* wb0 = wb + (size_t)nt0 * (KSTEPS * WBLK) + lane * 16;
    const bf16_t* wb1 = wb0 + (size_t)(KSTEPS * WBLK);
    if (p < 7) {                        // prefetch next pair's weight stream
      __builtin_prefetch(wb1 + (size_t)(KSTEPS * WBLK), 0, 0);
      __builtin_prefetch(wb1 + (size_t)(2 * KSTEPS * WBLK), 0, 0);
    }
    v8f acc00 = {};   // (rows m0..m0+15)  x nt0
    v8f acc01 = {};   // (rows m0..m0+15)  x nt1
    v8f acc10 = {};   // (rows m0+16..+31) x nt0
    v8f acc11 = {};   // (rows m0+16..+31) x nt1
    #pragma unroll 4
    for (int kt = 0; kt < KSTEPS; ++kt) {
      // A fragments 16x32: lanes0-15 K 0-15, lanes16-31 K 16-31
      const bf16_t* ap0 = A + (size_t)(m0 + llo) * XSTRIDE + kt * 32 + lhi * 16;
      const bf16_t* ap1 = ap0 + 16 * XSTRIDE;
      const v16bf afA = CAT16(*(const bf16x8*)ap0, *(const bf16x8*)(ap0 + 8));
      const v16bf afB = CAT16(*(const bf16x8*)ap1, *(const bf16x8*)(ap1 + 8));
      // B fragments: contiguous 32B per lane (pre-swizzled)
      const v16bf bf0 = CAT16(*(const bf16x8*)(wb0 + kt * WBLK),
                              *(const bf16x8*)(wb0 + kt * WBLK + 8));
      const v16bf bf1 = CAT16(*(const bf16x8*)(wb1 + kt * WBLK),
                              *(const bf16x8*)(wb1 + kt * WBLK + 8));
      acc00 = __builtin_amdgcn_wmma_f32_16x16x32_bf16(false, afA, false, bf0,
                                                      (short)0, acc00, false, false);
      acc01 = __builtin_amdgcn_wmma_f32_16x16x32_bf16(false, afA, false, bf1,
                                                      (short)0, acc01, false, false);
      acc10 = __builtin_amdgcn_wmma_f32_16x16x32_bf16(false, afB, false, bf0,
                                                      (short)0, acc10, false, false);
      acc11 = __builtin_amdgcn_wmma_f32_16x16x32_bf16(false, afB, false, bf1,
                                                      (short)0, acc11, false, false);
    }
    // epilogue: C/D layout -> elem r: row = base + r + 8*(lane>=16), col = lane&15
    const int nc0 = nt0 * 16 + llo;
    const int nc1 = nt1 * 16 + llo;
    const float bias0 = bias[nc0];
    const float bias1 = bias[nc1];
    #pragma unroll
    for (int r = 0; r < 8; ++r) {
      const int mrowA = m0 + r + lhi * 8;
      const int mrowB = mrowA + 16;
      if (EPI == 0) {
        float h00 = acc00[r] + bias0;  h00 = h00 > 0.f ? h00 : 0.f;
        float h01 = acc01[r] + bias1;  h01 = h01 > 0.f ? h01 : 0.f;
        float h10 = acc10[r] + bias0;  h10 = h10 > 0.f ? h10 : 0.f;
        float h11 = acc11[r] + bias1;  h11 = h11 > 0.f ? h11 : 0.f;
        sH[(size_t)mrowA * XSTRIDE + nc0] = (bf16_t)h00;
        sH[(size_t)mrowA * XSTRIDE + nc1] = (bf16_t)h01;
        sH[(size_t)mrowB * XSTRIDE + nc0] = (bf16_t)h10;
        sH[(size_t)mrowB * XSTRIDE + nc1] = (bf16_t)h11;
      } else {
        const int tokA = sTok[mrowA];
        if (tokA >= 0) {
          const float g = sGate[mrowA];
          unsafeAtomicAdd(out + (size_t)tokA * OUT_DIM + nc0, g * (acc00[r] + bias0));
          unsafeAtomicAdd(out + (size_t)tokA * OUT_DIM + nc1, g * (acc01[r] + bias1));
        }
        const int tokB = sTok[mrowB];
        if (tokB >= 0) {
          const float g = sGate[mrowB];
          unsafeAtomicAdd(out + (size_t)tokB * OUT_DIM + nc0, g * (acc10[r] + bias0));
          unsafeAtomicAdd(out + (size_t)tokB * OUT_DIM + nc1, g * (acc11[r] + bias1));
        }
      }
    }
  }
}

// ---------------- fused expert FFN: gather -> GEMM -> relu -> GEMM -> scatter ----------------
__global__ __launch_bounds__(256)
void moe_expert_kernel(const bf16_t* __restrict__ xbf,
                       const bf16_t* __restrict__ w1bf, const float* __restrict__ b1,
                       const bf16_t* __restrict__ w2bf, const float* __restrict__ b2,
                       const int* __restrict__ cnt, const int* __restrict__ lists,
                       const float* __restrict__ gates,
                       float* __restrict__ out)
{
  extern __shared__ char smem_raw[];
  bf16_t* sX    = (bf16_t*)smem_raw;                 // [M_TILE][XSTRIDE]
  bf16_t* sH    = sX + M_TILE * XSTRIDE;             // [M_TILE][XSTRIDE]
  int*    sTok  = (int*)(sH + M_TILE * XSTRIDE);
  float*  sGate = (float*)(sTok + M_TILE);

  const int e     = blockIdx.x / TILES_PER_E;
  const int tile  = blockIdx.x % TILES_PER_E;
  const int count = cnt[e];
  const int row0  = tile * M_TILE;
  if (row0 >= count) return;

  const int tid = threadIdx.x;
  if (tid < M_TILE) {
    const int r = row0 + tid;
    const bool ok = (r < count);
    sTok[tid]  = ok ? lists[e * NTOK + r] : -1;
    sGate[tid] = ok ? gates[e * NTOK + r] : 0.f;
  }
  __syncthreads();

  // gather token rows (already bf16) into LDS, 16B chunks
  for (int idx = tid; idx < M_TILE * (IN_DIM / 8); idx += 256) {
    const int r  = idx >> 7;            // row 0..63
    const int c8 = (idx & 127) * 8;     // col 0..1016
    const int tokv = sTok[r];
    bf16x8 v = {};
    if (tokv >= 0) v = *(const bf16x8*)(xbf + (size_t)tokv * IN_DIM + c8);
    *(bf16x8*)(sX + (size_t)r * XSTRIDE + c8) = v;
  }
  __syncthreads();

  ffn_layer<0>(sX, sH, sTok, sGate, w1bf + (size_t)e * IN_DIM * OUT_DIM,
               b1 + (size_t)e * OUT_DIM, nullptr);
  __syncthreads();   // sH complete before layer 2 reads it
  ffn_layer<1>(sH, sH, sTok, sGate, w2bf + (size_t)e * OUT_DIM * OUT_DIM,
               b2 + (size_t)e * OUT_DIM, out);
}

// ---------------- launcher ----------------
extern "C" void kernel_launch(void* const* d_in, const int* in_sizes, int n_in,
                              void* d_out, int out_size, void* d_ws, size_t ws_size,
                              hipStream_t stream)
{
  const float* x  = (const float*)d_in[0];
  const float* rw = (const float*)d_in[1];
  const float* rb = (const float*)d_in[2];
  const float* w1 = (const float*)d_in[3];
  const float* b1 = (const float*)d_in[4];
  const float* w2 = (const float*)d_in[5];
  const float* b2 = (const float*)d_in[6];

  float* fused  = (float*)d_out;
  float* gating = fused + (size_t)NTOK * OUT_DIM;

  char*   ws    = (char*)d_ws;
  int*    cnt   = (int*)ws;                                          // 8 ints
  int*    lists = (int*)(ws + 256);                                  // 8*8192 ints
  float*  gates = (float*)(ws + 256 + sizeof(int) * NUM_E * NTOK);   // 8*8192 f32
  size_t  off   = 256 + (size_t)2 * sizeof(int) * NUM_E * NTOK;      // 524544, 16B aligned
  bf16_t* xbf   = (bf16_t*)(ws + off);                               // 16 MB
  bf16_t* w1bf  = (bf16_t*)(ws + off + (size_t)NTOK * IN_DIM * 2);   // 16 MB
  bf16_t* w2bf  = w1bf + (size_t)NUM_E * IN_DIM * OUT_DIM;           // 16 MB

  const size_t smem_bytes = (size_t)2 * M_TILE * XSTRIDE * sizeof(bf16_t)
                          + M_TILE * (sizeof(int) + sizeof(float));  // ~264.7 KB

  moe_zero_kernel<<<1, 32, 0, stream>>>(cnt);
  moe_prep_w<<<dim3((NUM_E * NT_TILES * KSTEPS * 32) / 256, 2, 1), 256, 0, stream>>>(
      w1, w2, w1bf, w2bf);
  moe_router_kernel<<<(NTOK * 32) / 256, 256, 0, stream>>>(x, rw, rb, fused, gating,
                                                           xbf, cnt, lists, gates);
  moe_expert_kernel<<<NUM_E * TILES_PER_E, 256, smem_bytes, stream>>>(
      xbf, w1bf, b1, w2bf, b2, cnt, lists, gates, fused);
}
